// BalancedSkipGramModel_69526930588493
// MI455X (gfx1250) — matrix operand order repaired and checked by reading, not compiled, not run on hardware
//
#include <hip/hip_runtime.h>

// ---- reference problem constants --------------------------------------------
#define BB   256
#define LKV  75
#define KK   5
#define MM   5
#define DIMV 128
#define NWIN (BB*LKV)              // 19200 windows
#define NPOS (NWIN*KK)             // 96000
#define NNEG (NWIN*KK*MM)          // 480000
#define OFF_NEG   (NPOS)
#define OFF_PTPOS (NPOS+NNEG)
#define OFF_PTNEG (NPOS+NNEG+NPOS)
#define STRH 136                   // padded LDS row stride in halves (16B multiple)

typedef __attribute__((ext_vector_type(16))) _Float16 v16h;
typedef __attribute__((ext_vector_type(8)))  _Float16 v8h;
typedef __attribute__((ext_vector_type(4)))  _Float16 v4h;
typedef __attribute__((ext_vector_type(8)))  float    v8f;

__device__ __forceinline__ v16h cat8(v8h a, v8h b) {
  v16h r;
#pragma unroll
  for (int i = 0; i < 8; ++i) { r[i] = a[i]; r[i+8] = b[i]; }
  return r;
}

// A-matrix fragment (16x32 f16): lane m<16 rows; hi=0 -> K{0..7,16..23}, hi=1 -> K{8..15,24..31}
__device__ __forceinline__ v16h lds_frag_a(const _Float16* base, int m, int hi, int c) {
  const v8h r1 = *(const v8h*)(base + m*STRH + 32*c + 8*hi);
  const v8h r2 = *(const v8h*)(base + m*STRH + 32*c + 16 + 8*hi);
  return cat8(r1, r2);
}

__global__ __launch_bounds__(256)
void skipgram_wmma_kernel(const int*   __restrict__ walk,
                          const int*   __restrict__ pos,
                          const int*   __restrict__ neg,
                          const int*   __restrict__ walk_type,
                          const int*   __restrict__ pos_type,
                          const int*   __restrict__ neg_type,
                          const float* __restrict__ node_emb,
                          const float* __restrict__ rel_emb,
                          float*       __restrict__ out)
{
  __shared__ float    s_sig[16 * DIMV];        // sigmoid(rel) table, 8 KB
  __shared__ _Float16 s_walk[2][16 * STRH];    // 2 tiles x 16 walk rows (f16)
  __shared__ _Float16 s_ctx[8][16 * STRH];     // per-wave gated context tile (f16)

  const int tid  = threadIdx.x;
  const int wave = tid >> 5;
  const int lane = tid & 31;
  const int tile = wave >> 2;       // 0 or 1: which 16-window tile
  const int wit  = wave & 3;        // wave-in-tile: splits the 30 contexts
  const int windowBase = blockIdx.x * 32 + tile * 16;

  // ---- sigmoid(relationship) table: 2048 floats, 8 per thread ---------------
#pragma unroll
  for (int i = 0; i < 8; ++i) {
    const int idx = tid + i * 256;
    const float v = rel_emb[idx];
    s_sig[idx] = 1.0f / (1.0f + __expf(-v));
  }

  // ---- stage 16 walk embeddings for this tile (4 rows per wave) -------------
#pragma unroll
  for (int rr = 0; rr < 4; ++rr) {
    const int r = wit * 4 + rr;
    const int g = windowBase + r;
    const int wi = walk[g];
    const float4 e = *(const float4*)(node_emb + (size_t)wi * DIMV + lane * 4);
    v4h h;
    h[0] = (_Float16)e.x; h[1] = (_Float16)e.y;
    h[2] = (_Float16)e.z; h[3] = (_Float16)e.w;
    *(v4h*)(&s_walk[tile][r * STRH + lane * 4]) = h;
  }
  __syncthreads();   // sole workgroup barrier (uniform)

  const int m  = lane & 15;
  const int hi = lane >> 4;

  // A fragments (walk tile) are reused across all 30 contexts: hoist them.
  const v16h A0 = lds_frag_a(s_walk[tile], m, hi, 0);
  const v16h A1 = lds_frag_a(s_walk[tile], m, hi, 1);
  const v16h A2 = lds_frag_a(s_walk[tile], m, hi, 2);
  const v16h A3 = lds_frag_a(s_walk[tile], m, hi, 3);

  const int gLane = windowBase + (lane & 15);
  int wt = 0;
  if (lane < 16) wt = walk_type[gLane];

  // B fragment base: lane n = column (m), hi selects K{8hi..8hi+7,16+8hi..23+8hi}; rows stride STRH.
  const _Float16* cb = s_ctx[wave] + m * STRH + 8 * hi;

  // contexts j: 0..4 = positives (k=j), 5..29 = negatives (flat k*M+m)
  for (int j = wit; j < 30; j += 4) {
    const int isPos = (j < 5);
    const int jn = j - 5;
    const int k  = isPos ? j : (jn / MM);

    int cidx = 0, pair = 0;
    if (lane < 16) {
      int t;
      if (isPos) { cidx = pos[gLane * KK + j];          t = pos_type[gLane * KK + j]; }
      else       { cidx = neg[gLane * (KK*MM) + jn];    t = neg_type[gLane * (KK*MM) + jn]; }
      pair = 4 * wt + t;
      const float ptv = (float)(16 * k + pair);         // T*T*k + pair_type
      if (isPos) out[OFF_PTPOS + gLane * KK + j]       = ptv;
      else       out[OFF_PTNEG + gLane * (KK*MM) + jn] = ptv;
    }

    // ---- stage 16 gated context embeddings (f16) for this wave --------------
#pragma unroll 8
    for (int r = 0; r < 16; ++r) {
      const int si = __shfl(cidx, r);
      const int pr = __shfl(pair, r);
      const float4 e  = *(const float4*)(node_emb + (size_t)si * DIMV + lane * 4);
      const float4 gt = *(const float4*)(&s_sig[pr * DIMV + lane * 4]);
      v4h h;
      h[0] = (_Float16)(e.x * gt.x); h[1] = (_Float16)(e.y * gt.y);
      h[2] = (_Float16)(e.z * gt.z); h[3] = (_Float16)(e.w * gt.w);
      *(v4h*)(&s_ctx[wave][r * STRH + lane * 4]) = h;
    }
    // wave-local ordering of LDS stores vs. WMMA operand loads (no wg barrier:
    // waves execute different context counts). DS ops are in-order per wave.
    asm volatile("s_wait_dscnt 0" ::: "memory");

    // ---- issue ALL B-fragment loads first, then chain the 4 WMMAs -----------
    // (DS returns in order per wave -> compiler can use partial dscnt waits
    //  and overlap later loads with earlier WMMAs.)
    const v8h p0a = *(const v8h*)(cb +   0);
    const v8h p0b = *(const v8h*)(cb +  16);
    const v8h p1a = *(const v8h*)(cb +  32);
    const v8h p1b = *(const v8h*)(cb +  48);
    const v8h p2a = *(const v8h*)(cb +  64);
    const v8h p2b = *(const v8h*)(cb +  80);
    const v8h p3a = *(const v8h*)(cb +  96);
    const v8h p3b = *(const v8h*)(cb + 112);

    v8f acc = {};
    acc = __builtin_amdgcn_wmma_f32_16x16x32_f16(false, A0, false, cat8(p0a, p0b), (short)0, acc, false, false);
    acc = __builtin_amdgcn_wmma_f32_16x16x32_f16(false, A1, false, cat8(p1a, p1b), (short)0, acc, false, false);
    acc = __builtin_amdgcn_wmma_f32_16x16x32_f16(false, A2, false, cat8(p2a, p2b), (short)0, acc, false, false);
    acc = __builtin_amdgcn_wmma_f32_16x16x32_f16(false, A3, false, cat8(p3a, p3b), (short)0, acc, false, false);

    // Diagonal D[w][w]: lanes 0..7 hold w=lane in acc[lane];
    // lanes 24..31 hold w=lane-16 in acc[lane-24].
    const int e = (lane < 8) ? lane : (lane - 24);
    float sv = acc[0];
#pragma unroll
    for (int i = 1; i < 8; ++i) if (e == i) sv = acc[i];

    const bool doStore = (lane < 8) || (lane >= 24);
    if (doStore) {
      const int w = (lane < 8) ? lane : (lane - 16);
      const int g = windowBase + w;
      if (isPos) out[g * KK + j]                  = sv;
      else       out[OFF_NEG + g * (KK*MM) + jn]  = sv;
    }
  }
}

extern "C" void kernel_launch(void* const* d_in, const int* in_sizes, int n_in,
                              void* d_out, int out_size, void* d_ws, size_t ws_size,
                              hipStream_t stream) {
  (void)in_sizes; (void)n_in; (void)out_size; (void)d_ws; (void)ws_size;
  const int*   walk      = (const int*)  d_in[0];
  const int*   pos       = (const int*)  d_in[1];
  const int*   neg       = (const int*)  d_in[2];
  const int*   walk_type = (const int*)  d_in[3];
  const int*   pos_type  = (const int*)  d_in[4];
  const int*   neg_type  = (const int*)  d_in[5];
  const float* node_emb  = (const float*)d_in[6];
  const float* rel_emb   = (const float*)d_in[7];
  float*       out       = (float*)d_out;

  const dim3 grid(NWIN / 32);   // 600 blocks: 32 windows per 256-thread block
  const dim3 block(256);
  skipgram_wmma_kernel<<<grid, block, 0, stream>>>(
      walk, pos, neg, walk_type, pos_type, neg_type, node_emb, rel_emb, out);
}
